// KernelEncoder_75900662054930
// MI455X (gfx1250) — matrix operand order repaired and checked by reading, not compiled, not run on hardware
//
#include <hip/hip_runtime.h>

typedef __bf16 bh;
typedef __attribute__((ext_vector_type(16))) __bf16 v16bh;
typedef __attribute__((ext_vector_type(8)))  __bf16 bh8;
typedef __attribute__((ext_vector_type(8)))  float  v8f;

#define BN   32768      // B*N rows
#define NSEQ 8192       // N per batch
#define HM   2048       // H*M
#define HHD  1024       // H*HD

enum { M_QP = 0, M_KP_T = 1, M_V_T = 2, M_ATTNOUT = 3, M_FFN1 = 4, M_FFN2 = 5 };

// ---------------- WMMA fragment helpers (gfx1250 bf16 16x16x32 layouts) ----------

// A: 16x32 row-major tile. lane 0-15: row=lane, K=0..7 (e0..7), K=16..23 (e8..15)
//    lane 16-31: row=lane-16, K=8..15 (e0..7), K=24..31 (e8..15)
__device__ __forceinline__ v16bh load_A_frag(const bh* __restrict__ A, int lda,
                                             int row0, int k0, int lane) {
  int r  = row0 + (lane & 15);
  int kb = k0 + ((lane >> 4) << 3);
  const bh* p = A + (size_t)r * lda + kb;
  bh8 lo = *(const bh8*)p;
  bh8 hi = *(const bh8*)(p + 16);
  v16bh f;
#pragma unroll
  for (int i = 0; i < 8; ++i) { f[i] = lo[i]; f[i + 8] = hi[i]; }
  return f;
}

// B: 32x16 tile supplied TRANSPOSED in memory as Bt[n][K] (ld = ldb).
//    lane 0-15: col n=lane, K=0..7 (e0..7), K=16..23 (e8..15) ; lane 16-31: col n=lane-16, K=8..15 (e0..7), K=24..31 (e8..15)
__device__ __forceinline__ v16bh load_B_frag(const bh* __restrict__ Bt, int ldb,
                                             int n0, int k0, int lane) {
  int n  = n0 + (lane & 15);
  int kb = k0 + ((lane >> 4) << 3);
  const bh* p = Bt + (size_t)n * ldb + kb;
  bh8 lo = *(const bh8*)p;
  bh8 hi = *(const bh8*)(p + 16);
  v16bh f;
#pragma unroll
  for (int i = 0; i < 8; ++i) { f[i] = lo[i]; f[i + 8] = hi[i]; }
  return f;
}

__device__ __forceinline__ v8f wmma_bf16(v16bh a, v16bh b, v8f c) {
  return __builtin_amdgcn_wmma_f32_16x16x32_bf16(false, a, false, b, (short)0, c,
                                                 false, false);
}

__device__ __forceinline__ float wave_sum(float v) {
#pragma unroll
  for (int off = 16; off > 0; off >>= 1) v += __shfl_xor(v, off, 32);
  return v;
}

__device__ __forceinline__ float phi_f(float x) {           // (elu(x)+1)/sqrt(256)
  return (x > 0.f ? x + 1.f : __expf(x)) * 0.0625f;
}

// ---------------- generic GEMM: C[32768 x Ncols] = A[.,K] * Bt[Ncols,K]^T -------
// wave computes 16 rows x 64 cols. Epilogue selected by mode.
__global__ __launch_bounds__(128) void gemm_kernel(
    const bh* __restrict__ A, int lda, const bh* __restrict__ Bt, int ldb, int K,
    const float* __restrict__ bias, void* __restrict__ outp, int Ncols, int mode,
    const int* __restrict__ maskI, const float* __restrict__ resid) {
  const int lane = threadIdx.x & 31;
  const int w    = blockIdx.x * (blockDim.x >> 5) + (threadIdx.x >> 5);
  const int colStrips = Ncols >> 6;
  const int rt = w / colStrips, cs = w % colStrips;
  const int row0 = rt << 4, col0 = cs << 6;

  v8f acc[4];
#pragma unroll
  for (int c = 0; c < 4; ++c)
#pragma unroll
    for (int i = 0; i < 8; ++i) acc[c][i] = 0.f;

  for (int k0 = 0; k0 < K; k0 += 32) {
    v16bh a = load_A_frag(A, lda, row0, k0, lane);
#pragma unroll
    for (int c = 0; c < 4; ++c) {
      v16bh b = load_B_frag(Bt, ldb, col0 + (c << 4), k0, lane);
      acc[c]  = wmma_bf16(a, b, acc[c]);
    }
  }

  const int lanehi = lane >> 4, lncol = lane & 15;
#pragma unroll
  for (int c = 0; c < 4; ++c) {
    int   n   = col0 + (c << 4) + lncol;
    float bia = bias ? bias[n] : 0.f;
    if (mode == M_KP_T || mode == M_V_T) {
      // store transposed: out[(b*Ncols + n) * 8192 + t], 8 consecutive t per lane
      int b  = row0 >> 13;
      int t0 = (row0 & (NSEQ - 1)) + (lanehi << 3);
      bh8 pk;
#pragma unroll
      for (int v = 0; v < 8; ++v) {
        int   rowg = row0 + v + (lanehi << 3);
        float x    = acc[c][v] + bia;
        if (mode == M_KP_T) x = phi_f(x) * (float)maskI[rowg];
        pk[v] = (bh)x;
      }
      bh* ob = (bh*)outp;
      *(bh8*)(ob + ((size_t)b * Ncols + n) * NSEQ + t0) = pk;
    } else {
#pragma unroll
      for (int v = 0; v < 8; ++v) {
        int   rowg = row0 + v + (lanehi << 3);
        float x    = acc[c][v] + bia;
        if (mode == M_QP)           x = phi_f(x);
        else if (mode == M_FFN1)    x = (x > 0.f ? x : __expf(x) - 1.f);
        else if (mode == M_ATTNOUT) x = x * (float)maskI[rowg] +
                                        resid[(size_t)rowg * Ncols + n];
        if (mode == M_ATTNOUT || mode == M_FFN2)
          ((float*)outp)[(size_t)rowg * Ncols + n] = x;
        else
          ((bh*)outp)[(size_t)rowg * Ncols + n] = (bh)x;
      }
    }
  }
}

// ---------------- kv[b,h,m,d] = sum_t kpT[bh,m,t] * v[b,t,h,d]  (K split x8) ----
__global__ __launch_bounds__(32) void kv_kernel(const bh* __restrict__ kpT,
                                                const bh* __restrict__ vT,
                                                float* __restrict__ kvf) {
  int lane = threadIdx.x;
  int blk  = blockIdx.x;           // bh(32) * mtile(16) * kchunk(8)
  int kc = blk & 7, mt = (blk >> 3) & 15, bhid = blk >> 7;
  int b = bhid >> 3, h = bhid & 7;
  const bh* Ab = kpT + (size_t)bhid * 256 * NSEQ;                 // [m][t]
  const bh* Bb = vT + ((size_t)b * HHD + h * 128) * NSEQ;         // [d][t]
  int row0 = mt << 4;
  int kbeg = kc << 10, kend = kbeg + 1024;

  v8f acc[8];
#pragma unroll
  for (int c = 0; c < 8; ++c)
#pragma unroll
    for (int i = 0; i < 8; ++i) acc[c][i] = 0.f;

  for (int k0 = kbeg; k0 < kend; k0 += 32) {
    v16bh a = load_A_frag(Ab, NSEQ, row0, k0, lane);
#pragma unroll
    for (int c = 0; c < 8; ++c) {
      v16bh bfr = load_B_frag(Bb, NSEQ, c << 4, k0, lane);
      acc[c]    = wmma_bf16(a, bfr, acc[c]);
    }
  }
  int    lanehi = lane >> 4, ln = lane & 15;
  float* ob = kvf + (size_t)bhid * 256 * 128;
#pragma unroll
  for (int c = 0; c < 8; ++c) {
    int d = (c << 4) + ln;
#pragma unroll
    for (int v = 0; v < 8; ++v) {
      int m = row0 + v + (lanehi << 3);
      atomicAdd(&ob[m * 128 + d], acc[c][v]);
    }
  }
}

// ---------------- attn[t, h*128+d] = den[t,h] * sum_m qp[t,h*256+m]*kv[b,h,m,d] -
__global__ __launch_bounds__(32) void attn_kernel(const bh* __restrict__ qp,
                                                  const bh* __restrict__ kvT,
                                                  const float* __restrict__ den,
                                                  bh* __restrict__ attnb) {
  int lane = threadIdx.x;
  int blk  = blockIdx.x;           // rowtile(2048) * H(8)
  int h = blk & 7, rt = blk >> 3;
  int row0 = rt << 4;
  int b    = row0 >> 13;
  const bh* Ab = qp + (size_t)row0 * HM + h * 256;                // lda = HM
  const bh* Bb = kvT + ((size_t)(b * 8 + h)) * 128 * 256;         // [d][m]

  v8f acc[8];
#pragma unroll
  for (int c = 0; c < 8; ++c)
#pragma unroll
    for (int i = 0; i < 8; ++i) acc[c][i] = 0.f;

  for (int k0 = 0; k0 < 256; k0 += 32) {
    v16bh a = load_A_frag(Ab, HM, 0, k0, lane);
#pragma unroll
    for (int c = 0; c < 8; ++c) {
      v16bh bfr = load_B_frag(Bb, 256, c << 4, k0, lane);
      acc[c]    = wmma_bf16(a, bfr, acc[c]);
    }
  }
  int   lanehi = lane >> 4, ln = lane & 15;
  float dv[8];
#pragma unroll
  for (int v = 0; v < 8; ++v)
    dv[v] = den[(size_t)(row0 + v + (lanehi << 3)) * 8 + h];
#pragma unroll
  for (int c = 0; c < 8; ++c) {
    int d = (c << 4) + ln;
#pragma unroll
    for (int v = 0; v < 8; ++v) {
      int t = row0 + v + (lanehi << 3);
      attnb[(size_t)t * HHD + h * 128 + d] = (bh)(acc[c][v] * dv[v]);
    }
  }
}

// ---------------- small VALU kernels ---------------------------------------------
__global__ void ln_kernel(const float* __restrict__ X, const float* __restrict__ g,
                          const float* __restrict__ be, float* __restrict__ outf,
                          bh* __restrict__ outb) {
  int w    = (blockIdx.x * blockDim.x + threadIdx.x) >> 5;   // row
  int lane = threadIdx.x & 31;
  const float* xr = X + (size_t)w * 128;
  float4 x = *(const float4*)(xr + lane * 4);
  float  s  = x.x + x.y + x.z + x.w;
  float  s2 = x.x * x.x + x.y * x.y + x.z * x.z + x.w * x.w;
  s  = wave_sum(s);
  s2 = wave_sum(s2);
  float mu  = s * (1.f / 128.f);
  float var = s2 * (1.f / 128.f) - mu * mu;
  float sc  = rsqrtf(var + 1e-3f);
  float xi[4] = {x.x, x.y, x.z, x.w};
#pragma unroll
  for (int i = 0; i < 4; ++i) {
    int   c = lane * 4 + i;
    float y = (xi[i] - mu) * sc * g[c] + be[c];
    if (outf) outf[(size_t)w * 128 + c] = y;
    outb[(size_t)w * 128 + c] = (bh)y;
  }
}

__global__ void cvt_kernel(const float* __restrict__ in, bh* __restrict__ out, int n) {
  int i = blockIdx.x * blockDim.x + threadIdx.x;
  if (i < n) out[i] = (bh)in[i];
}

__global__ void zero_kernel(float* __restrict__ p, int n) {
  int i = blockIdx.x * blockDim.x + threadIdx.x;
  if (i < n) p[i] = 0.f;
}

__global__ void ksum_kernel(const bh* __restrict__ kpT, float* __restrict__ ksum) {
  int w    = (blockIdx.x * blockDim.x + threadIdx.x) >> 5;   // b*2048+h*256+m
  int lane = threadIdx.x & 31;
  const bh* r = kpT + (size_t)w * NSEQ;
  float s = 0.f;
  for (int it = 0; it < 32; ++it) {
    bh8 v = *(const bh8*)(r + it * 256 + lane * 8);
#pragma unroll
    for (int j = 0; j < 8; ++j) s += (float)v[j];
  }
  s = wave_sum(s);
  if (lane == 0) ksum[w] = s;
}

__global__ void den_kernel(const bh* __restrict__ qp, const float* __restrict__ ksum,
                           float* __restrict__ den) {
  int w    = (blockIdx.x * blockDim.x + threadIdx.x) >> 5;   // row*8+h
  int lane = threadIdx.x & 31;
  int row = w >> 3, h = w & 7, b = row >> 13;
  bh8 qv = *(const bh8*)(qp + (size_t)row * HM + h * 256 + lane * 8);
  const float* ks = ksum + (size_t)(b * 8 + h) * 256 + lane * 8;
  float s = 0.f;
#pragma unroll
  for (int j = 0; j < 8; ++j) s += (float)qv[j] * ks[j];
  s = wave_sum(s);
  if (lane == 0) den[w] = 1.f / (s + 1e-6f);
}

__global__ void kvt_kernel(const float* __restrict__ kvf, bh* __restrict__ kvT, int n) {
  int i = blockIdx.x * blockDim.x + threadIdx.x;
  if (i < n) {
    int d = i & 127, m = (i >> 7) & 255, bhid = i >> 15;
    kvT[((size_t)bhid * 128 + d) * 256 + m] = (bh)kvf[i];
  }
}

// fold proj into Wq/Wk (AqT = (Wq_h @ proj^T)^T etc.), transpose remaining weights
__global__ void prep_kernel(const float* __restrict__ Wq, const float* __restrict__ bq,
                            const float* __restrict__ Wk, const float* __restrict__ bk,
                            const float* __restrict__ Wv, const float* __restrict__ Wo,
                            const float* __restrict__ W1, const float* __restrict__ W2,
                            const float* __restrict__ proj, bh* AqT, bh* AkT,
                            float* cq, float* ck, bh* WvT, bh* WoT, bh* W1T, bh* W2T) {
  int id = blockIdx.x * blockDim.x + threadIdx.x;
  if (id < 262144) {                 // AqT[n*128+d], n=h*256+m
    int n = id >> 7, d = id & 127, h = n >> 8, m = n & 255;
    const float* wp = Wq + (size_t)d * HHD + h * 128;
    const float* pp = proj + (size_t)m * 128;
    float s = 0.f;
    for (int j = 0; j < 128; ++j) s += wp[j] * pp[j];
    AqT[id] = (bh)s; return;
  }
  id -= 262144;
  if (id < 262144) {
    int n = id >> 7, d = id & 127, h = n >> 8, m = n & 255;
    const float* wp = Wk + (size_t)d * HHD + h * 128;
    const float* pp = proj + (size_t)m * 128;
    float s = 0.f;
    for (int j = 0; j < 128; ++j) s += wp[j] * pp[j];
    AkT[id] = (bh)s; return;
  }
  id -= 262144;
  if (id < 2048) {                   // cq[n] = bq_h . proj_m
    int h = id >> 8, m = id & 255;
    const float* bp = bq + h * 128;
    const float* pp = proj + (size_t)m * 128;
    float s = 0.f;
    for (int j = 0; j < 128; ++j) s += bp[j] * pp[j];
    cq[id] = s; return;
  }
  id -= 2048;
  if (id < 2048) {
    int h = id >> 8, m = id & 255;
    const float* bp = bk + h * 128;
    const float* pp = proj + (size_t)m * 128;
    float s = 0.f;
    for (int j = 0; j < 128; ++j) s += bp[j] * pp[j];
    ck[id] = s; return;
  }
  id -= 2048;
  if (id < 131072) { int n = id >> 7, k = id & 127; WvT[id] = (bh)Wv[(size_t)k * HHD + n]; return; }
  id -= 131072;
  if (id < 131072) { int n = id >> 10, kk = id & 1023; WoT[id] = (bh)Wo[(size_t)kk * 128 + n]; return; }
  id -= 131072;
  if (id < 16384) { int n = id >> 7, k = id & 127; W1T[id] = (bh)W1[(size_t)k * 128 + n]; return; }
  id -= 16384;
  if (id < 16384) { int n = id >> 7, k = id & 127; W2T[id] = (bh)W2[(size_t)k * 128 + n]; return; }
}

// ---------------- host launcher --------------------------------------------------
extern "C" void kernel_launch(void* const* d_in, const int* in_sizes, int n_in,
                              void* d_out, int out_size, void* d_ws, size_t ws_size,
                              hipStream_t stream) {
  const float* Q    = (const float*)d_in[0];
  const float* X    = (const float*)d_in[1];
  const int*   mask = (const int*)d_in[2];
  const float* Wq   = (const float*)d_in[3];
  const float* bq   = (const float*)d_in[4];
  const float* Wk   = (const float*)d_in[5];
  const float* bk   = (const float*)d_in[6];
  const float* Wv   = (const float*)d_in[7];
  const float* bv   = (const float*)d_in[8];
  const float* Wo   = (const float*)d_in[9];
  const float* bo   = (const float*)d_in[10];
  const float* proj = (const float*)d_in[11];
  const float* ln1g = (const float*)d_in[12];
  const float* ln1b = (const float*)d_in[13];
  const float* ln2g = (const float*)d_in[14];
  const float* ln2b = (const float*)d_in[15];
  const float* W1   = (const float*)d_in[16];
  const float* b1   = (const float*)d_in[17];
  const float* W2   = (const float*)d_in[18];
  const float* b2   = (const float*)d_in[19];

  char*  ws  = (char*)d_ws;
  size_t off = 0;
  auto alloc = [&](size_t bytes) -> char* {
    char* p = ws + off;
    off += (bytes + 255) & ~(size_t)255;
    return p;
  };
  float* Xn    = (float*)alloc((size_t)BN * 128 * 4);
  bh*    Xnb   = (bh*)alloc((size_t)BN * 128 * 2);
  bh*    Qb    = (bh*)alloc((size_t)BN * 128 * 2);
  bh*    qp    = (bh*)alloc((size_t)BN * HM * 2);
  char*  kpreg = alloc((size_t)BN * HM * 2);
  bh*    kpT   = (bh*)kpreg;                                   // consumed by ksum/kv
  bh*    attnb = (bh*)kpreg;                                   // reuse after kv
  float* rbuf  = (float*)(kpreg + (size_t)BN * HHD * 2);
  bh*    out1b = (bh*)(kpreg + (size_t)BN * HHD * 2 + (size_t)BN * 128 * 4);
  bh*    hb    = (bh*)((char*)out1b + (size_t)BN * 128 * 2);
  bh*    vT    = (bh*)alloc((size_t)BN * HHD * 2);
  float* ksum  = (float*)alloc(8192 * 4);
  float* kvf   = (float*)alloc((size_t)32 * 256 * 128 * 4);
  bh*    kvT   = (bh*)alloc((size_t)32 * 256 * 128 * 2);
  float* den   = (float*)alloc((size_t)BN * 8 * 4);
  bh*    AqT   = (bh*)alloc((size_t)HM * 128 * 2);
  bh*    AkT   = (bh*)alloc((size_t)HM * 128 * 2);
  float* cq    = (float*)alloc(HM * 4);
  float* ck    = (float*)alloc(HM * 4);
  bh*    WvT   = (bh*)alloc((size_t)HHD * 128 * 2);
  bh*    WoT   = (bh*)alloc((size_t)128 * HHD * 2);
  bh*    W1T   = (bh*)alloc((size_t)128 * 128 * 2);
  bh*    W2T   = (bh*)alloc((size_t)128 * 128 * 2);

  // 1. fold proj into Wq/Wk, transpose weights to [N][K] bf16
  prep_kernel<<<3217, 256, 0, stream>>>(Wq, bq, Wk, bk, Wv, Wo, W1, W2, proj,
                                        AqT, AkT, cq, ck, WvT, WoT, W1T, W2T);
  // 2. LN1 (f32 + bf16), Q -> bf16
  ln_kernel<<<4096, 256, 0, stream>>>(X, ln1g, ln1b, Xn, Xnb);
  cvt_kernel<<<16384, 256, 0, stream>>>(Q, Qb, BN * 128);
  // 3. qp = phi(Q @ AqT^T + cq)                [32768 x 2048] bf16
  gemm_kernel<<<16384, 128, 0, stream>>>(Qb, 128, AqT, 128, 128, cq, qp, HM,
                                         M_QP, nullptr, nullptr);
  // 4. kpT = (phi(Xn @ AkT^T + ck) * mask)^T   [b,h,m][t] bf16
  gemm_kernel<<<16384, 128, 0, stream>>>(Xnb, 128, AkT, 128, 128, ck, kpT, HM,
                                         M_KP_T, mask, nullptr);
  // 5. vT = (Xn @ Wv + bv)^T                   [b,h,d][t] bf16
  gemm_kernel<<<8192, 128, 0, stream>>>(Xnb, 128, WvT, 128, 128, bv, vT, HHD,
                                        M_V_T, nullptr, nullptr);
  // 6. ksum[b,h,m] = sum_t kpT
  ksum_kernel<<<1024, 256, 0, stream>>>(kpT, ksum);
  // 7. kv = kpT @ v (K=8192, split x8 with f32 atomics), then -> kvT bf16
  zero_kernel<<<4096, 256, 0, stream>>>(kvf, 32 * 256 * 128);
  kv_kernel<<<4096, 32, 0, stream>>>(kpT, vT, kvf);
  kvt_kernel<<<4096, 256, 0, stream>>>(kvf, kvT, 32 * 256 * 128);
  // 8. den = 1/(qp . ksum + 1e-6)
  den_kernel<<<32768, 256, 0, stream>>>(qp, ksum, den);
  // 9. attn = (qp @ kv) * den                  [32768 x 1024] bf16 (reuses kpT mem)
  attn_kernel<<<16384, 32, 0, stream>>>(qp, kvT, den, attnb);
  // 10. r = (attn @ Wo + bo) * mask + Xn       f32
  gemm_kernel<<<1024, 128, 0, stream>>>(attnb, HHD, WoT, HHD, HHD, bo, rbuf, 128,
                                        M_ATTNOUT, mask, Xn);
  // 11. LN2 -> bf16
  ln_kernel<<<4096, 256, 0, stream>>>(rbuf, ln2g, ln2b, nullptr, out1b);
  // 12. FFN
  gemm_kernel<<<1024, 128, 0, stream>>>(out1b, 128, W1T, 128, 128, b1, hb, 128,
                                        M_FFN1, nullptr, nullptr);
  gemm_kernel<<<1024, 128, 0, stream>>>(hb, 128, W2T, 128, 128, b2, d_out, 128,
                                        M_FFN2, nullptr, nullptr);
}
